// MoEMLP_55783035240607
// MI455X (gfx1250) — compile-verified, matrix operand order checked by reading, not run on hardware
//
#include <hip/hip_runtime.h>
#include <math.h>

typedef __bf16 bf16_t;
typedef __attribute__((ext_vector_type(16))) __bf16 v16bf;
typedef __attribute__((ext_vector_type(8)))  float  v8f;

#define D_MODEL   1024
#define D_HIDDEN  4096
#define N_EXPERTS 8
#define TOP_K     2
#define N_TOKENS  8192
#define CAPACITY  1280   // ceil(8192/8 * 1.25)

// ---------------------------------------------------------------------------
// Tensor Data Mover: 2D bf16 tile load (global -> LDS), per CDNA5 ISA D#
// layout (group0: count/lds_addr/global_addr/type, group1: data_size, dims,
// tile dims, strides).  Tracked by TENSORcnt.
// ---------------------------------------------------------------------------
#if __has_builtin(__builtin_amdgcn_tensor_load_to_lds)
#define HAVE_TDM 1
typedef unsigned int u32x4 __attribute__((ext_vector_type(4)));
typedef int          i32x4 __attribute__((ext_vector_type(4)));
typedef int          i32x8 __attribute__((ext_vector_type(8)));

__device__ __forceinline__ void tdm_load_tile_bf16(
    unsigned lds_addr, const void* gaddr,
    unsigned tile_d0, unsigned tile_d1,          // elements / rows
    unsigned tensor_d0, unsigned tensor_d1,      // bounds (elements)
    unsigned long long stride_d0)                // row stride (elements)
{
    unsigned long long ga = (unsigned long long)gaddr;
    u32x4 g0;
    g0[0] = 1u;                                       // count=1, user D#
    g0[1] = lds_addr;                                 // LDS byte address
    g0[2] = (unsigned)(ga & 0xFFFFFFFFu);             // global_addr[31:0]
    g0[3] = (unsigned)((ga >> 32) & 0x01FFFFFFu) | (2u << 30);  // type=2
    i32x8 g1;
    g1[0] = (int)(1u << 16);                          // data_size=1 (2B), mask=0
    g1[1] = (int)((tensor_d0 & 0xFFFFu) << 16);       // tensor_dim0[15:0]
    g1[2] = (int)(((tensor_d0 >> 16) & 0xFFFFu) | ((tensor_d1 & 0xFFFFu) << 16));
    g1[3] = (int)(((tensor_d1 >> 16) & 0xFFFFu) | ((tile_d0 & 0xFFFFu) << 16));
    g1[4] = (int)(tile_d1 & 0xFFFFu);                 // tile_dim1, tile_dim2=0
    g1[5] = (int)(unsigned)(stride_d0 & 0xFFFFFFFFull);
    g1[6] = (int)(unsigned)((stride_d0 >> 32) & 0xFFFFull);
    g1[7] = 0;
    i32x4 z4 = {0, 0, 0, 0};
#if __clang_major__ >= 23
    i32x8 z8 = {0, 0, 0, 0, 0, 0, 0, 0};
    __builtin_amdgcn_tensor_load_to_lds(g0, g1, z4, z4, z8, 0);
#else
    __builtin_amdgcn_tensor_load_to_lds(g0, g1, z4, z4, 0);
#endif
}
#else
#define HAVE_TDM 0
#endif

// ---------------------------------------------------------------------------
// fp32 -> bf16 conversion / zero-fill (grid-stride)
// ---------------------------------------------------------------------------
__global__ void cvt_bf16_kernel(const float* __restrict__ src,
                                bf16_t* __restrict__ dst, size_t n) {
    size_t i = (size_t)blockIdx.x * blockDim.x + threadIdx.x;
    size_t stride = (size_t)gridDim.x * blockDim.x;
    for (; i < n; i += stride) dst[i] = (bf16_t)src[i];
}

__global__ void zero_u32_kernel(unsigned* __restrict__ p, size_t n) {
    size_t i = (size_t)blockIdx.x * blockDim.x + threadIdx.x;
    size_t stride = (size_t)gridDim.x * blockDim.x;
    for (; i < n; i += stride) p[i] = 0u;
}

// ---------------------------------------------------------------------------
// Router: logits = x @ router_w + b, softmax, top-2.  One wave32 per token.
// ---------------------------------------------------------------------------
__global__ __launch_bounds__(256) void router_kernel(
    const float* __restrict__ x, const float* __restrict__ rw,
    const float* __restrict__ rb, float* __restrict__ gates,
    int* __restrict__ topk) {
    int wave = threadIdx.x >> 5;
    int lane = threadIdx.x & 31;
    int n = blockIdx.x * 8 + wave;
    if (n >= N_TOKENS) return;

    float acc[N_EXPERTS];
#pragma unroll
    for (int e = 0; e < N_EXPERTS; e++) acc[e] = 0.f;
    const float* xr = x + (size_t)n * D_MODEL;
    for (int d = lane; d < D_MODEL; d += 32) {
        float xv = xr[d];
#pragma unroll
        for (int e = 0; e < N_EXPERTS; e++) acc[e] += xv * rw[d * N_EXPERTS + e];
    }
#pragma unroll
    for (int off = 16; off > 0; off >>= 1)
#pragma unroll
        for (int e = 0; e < N_EXPERTS; e++) acc[e] += __shfl_xor(acc[e], off, 32);

    if (lane == 0) {
        float mx = -1e30f;
#pragma unroll
        for (int e = 0; e < N_EXPERTS; e++) { acc[e] += rb[e]; mx = fmaxf(mx, acc[e]); }
        float s = 0.f;
#pragma unroll
        for (int e = 0; e < N_EXPERTS; e++) { acc[e] = __expf(acc[e] - mx); s += acc[e]; }
        float inv = 1.f / s;
#pragma unroll
        for (int e = 0; e < N_EXPERTS; e++) {
            acc[e] *= inv;
            gates[(size_t)n * N_EXPERTS + e] = acc[e];
        }
        int i0 = 0;
#pragma unroll
        for (int e = 1; e < N_EXPERTS; e++) if (acc[e] > acc[i0]) i0 = e;
        int i1 = (i0 == 0) ? 1 : 0;
#pragma unroll
        for (int e = 0; e < N_EXPERTS; e++)
            if (e != i0 && acc[e] > acc[i1]) i1 = e;
        topk[n * 2 + 0] = i0;
        topk[n * 2 + 1] = i1;
    }
}

// ---------------------------------------------------------------------------
// Capacity scan: one block, wave e scans all tokens for expert e using
// wave32 ballot + popcount prefix (== sequential cumsum semantics).
// ---------------------------------------------------------------------------
__global__ __launch_bounds__(256) void scan_kernel(
    const int* __restrict__ topk, const float* __restrict__ gates,
    int* __restrict__ pos, float* __restrict__ wgt) {
    int e = threadIdx.x >> 5;
    int lane = threadIdx.x & 31;
    int count = 0;
    for (int base = 0; base < N_TOKENS; base += 32) {
        int n = base + lane;
        int t0 = topk[n * 2], t1 = topk[n * 2 + 1];
        bool hit = (t0 == e) || (t1 == e);
        unsigned m = (unsigned)__ballot(hit);
        int prefix = __popc(m & ((1u << lane) - 1u));
        if (hit) {
            int p = count + prefix;
            int slot = (t0 == e) ? 0 : 1;
            pos[n * 2 + slot] = p;
            wgt[n * 2 + slot] = (p < CAPACITY) ? gates[(size_t)n * N_EXPERTS + e] : 0.f;
        }
        count += __popc(m);
    }
}

// ---------------------------------------------------------------------------
// Dispatch token rows into per-expert bf16 buffers
// ---------------------------------------------------------------------------
__global__ __launch_bounds__(256) void dispatch_kernel(
    const float* __restrict__ x, const int* __restrict__ topk,
    const int* __restrict__ pos, bf16_t* __restrict__ buf) {
    int slot = blockIdx.x;          // [0, N_TOKENS*TOP_K)
    int n = slot >> 1;
    int p = pos[slot];
    if (p >= CAPACITY) return;      // dropped (over capacity)
    int e = topk[slot];
    const float* src = x + (size_t)n * D_MODEL;
    bf16_t* dst = buf + ((size_t)e * CAPACITY + p) * D_MODEL;
    for (int d = threadIdx.x; d < D_MODEL; d += blockDim.x)
        dst[d] = (bf16_t)src[d];
}

// ---------------------------------------------------------------------------
// Batched expert GEMM: C[e] = act(A[e] @ B[e] + bias[e])
//   A: [E][CAPACITY][KDIM] bf16 (row major)
//   B: [E][KDIM][Ndim]     bf16 (row major)
// 128x128 tile per block (8 waves), K-step 32, double-buffered LDS.
// A-tile staged by the Tensor Data Mover (TENSORcnt), B-tile staged by
// vector loads + LDS transpose, fragments fed to v_wmma_f32_16x16x32_bf16.
// ---------------------------------------------------------------------------
template <int KDIM, bool DO_GELU, typename OUT_T>
__global__ __launch_bounds__(256) void moe_gemm_kernel(
    const bf16_t* __restrict__ A, const bf16_t* __restrict__ B,
    const float* __restrict__ bias, OUT_T* __restrict__ C, int Ndim) {
    constexpr int MT = 128, NT = 128, KT = 32;
    constexpr int NSTEP = KDIM / KT;
    __shared__ __align__(16) bf16_t lA[2][MT * KT];  // [m][k]
    __shared__ __align__(16) bf16_t lB[2][NT * KT];  // [n][k] (transposed)

    const int e  = blockIdx.z;
    const int m0 = blockIdx.y * MT;
    const int n0 = blockIdx.x * NT;
    const bf16_t* Ae = A + (size_t)e * CAPACITY * KDIM;
    const bf16_t* Be = B + (size_t)e * KDIM * Ndim;

    const int wave = threadIdx.x >> 5;
    const int lane = threadIdx.x & 31;
    const int kb   = (lane < 16) ? 0 : 8;   // K-half select per CDNA5 layout
    const int rc   = lane & 15;             // row (A) / col (B) within 16

    // B staging coordinates for this thread (32 k-rows x 128 cols)
    const int bk  = threadIdx.x >> 3;
    const int bc0 = (threadIdx.x & 7) * 16;

    v8f acc[8];
#pragma unroll
    for (int nf = 0; nf < 8; nf++)
#pragma unroll
        for (int i = 0; i < 8; i++) acc[nf][i] = 0.f;

    // ---- tile staging helpers -------------------------------------------
    auto stage_A = [&](int step, int buf) {
#if HAVE_TDM
        if (wave == 0) {
            tdm_load_tile_bf16((unsigned)(size_t)&lA[buf][0],
                               Ae + (size_t)m0 * KDIM + step * KT,
                               /*tile*/ KT, MT,
                               /*tensor*/ KDIM, CAPACITY,
                               /*stride*/ (unsigned long long)KDIM);
        }
#else
        int row = threadIdx.x >> 1;
        int off = (threadIdx.x & 1) * 16;
        const uint4* gp =
            (const uint4*)(Ae + (size_t)(m0 + row) * KDIM + step * KT + off);
        uint4 v0 = gp[0], v1 = gp[1];
        *(uint4*)&lA[buf][row * KT + off]     = v0;
        *(uint4*)&lA[buf][row * KT + off + 8] = v1;
#endif
    };
    auto load_B_global = [&](int step, uint4& v0, uint4& v1) {
        const bf16_t* gp = Be + (size_t)(step * KT + bk) * Ndim + n0 + bc0;
        v0 = *(const uint4*)gp;
        v1 = *(const uint4*)(gp + 8);
    };
    auto store_B_lds = [&](int buf, uint4 v0, uint4 v1) {
        const bf16_t* t0 = (const bf16_t*)&v0;
        const bf16_t* t1 = (const bf16_t*)&v1;
#pragma unroll
        for (int j = 0; j < 8; j++) lB[buf][(bc0 + j) * KT + bk] = t0[j];
#pragma unroll
        for (int j = 0; j < 8; j++) lB[buf][(bc0 + 8 + j) * KT + bk] = t1[j];
    };

    // ---- prologue: stage step 0 into buffer 0 ----------------------------
    stage_A(0, 0);
    {
        uint4 v0, v1;
        load_B_global(0, v0, v1);
        store_B_lds(0, v0, v1);
    }
#if HAVE_TDM
    __builtin_amdgcn_s_wait_tensorcnt(0);
#endif
    __syncthreads();

    int cb = 0;
    for (int s = 0; s < NSTEP; ++s) {
        const int nb = cb ^ 1;
        uint4 bv0, bv1;
        if (s + 1 < NSTEP) {
            stage_A(s + 1, nb);                 // async TDM into next buffer
            load_B_global(s + 1, bv0, bv1);
            if (s + 2 < NSTEP)                  // L2 prefetch of weight tile s+2
                __builtin_prefetch(Be + (size_t)((s + 2) * KT + bk) * Ndim + n0 + bc0,
                                   0, 1);
        }

        // ---- compute from buffer cb: gather all fragments, then WMMA chain
        v16bf a;
#pragma unroll
        for (int v = 0; v < 8; v++) {
            int kk = (v < 4) ? (kb + 2 * v) : (16 + kb + 2 * (v - 4));
            a[2 * v]     = lA[cb][(wave * 16 + rc) * KT + kk];
            a[2 * v + 1] = lA[cb][(wave * 16 + rc) * KT + kk + 1];
        }
        v16bf bf[8];
#pragma unroll
        for (int nf = 0; nf < 8; nf++) {
#pragma unroll
            for (int v = 0; v < 8; v++) {
                int kk = (v < 4) ? (kb + 2 * v) : (16 + kb + 2 * (v - 4));
                bf[nf][2 * v]     = lB[cb][(nf * 16 + rc) * KT + kk];
                bf[nf][2 * v + 1] = lB[cb][(nf * 16 + rc) * KT + kk + 1];
            }
        }
#pragma unroll
        for (int nf = 0; nf < 8; nf++)
            acc[nf] = __builtin_amdgcn_wmma_f32_16x16x32_bf16(
                false, a, false, bf[nf], (short)0, acc[nf], false, false);

        if (s + 1 < NSTEP) {
            store_B_lds(nb, bv0, bv1);
#if HAVE_TDM
            __builtin_amdgcn_s_wait_tensorcnt(0);
#endif
        }
        __syncthreads();
        cb = nb;
    }

    // ---- epilogue: bias (+ exact GELU), per CDNA5 16x16 f32 C layout -----
    const float* be = bias + (size_t)e * Ndim;
#pragma unroll
    for (int nf = 0; nf < 8; nf++) {
#pragma unroll
        for (int i = 0; i < 8; i++) {
            int row = m0 + wave * 16 + i + ((lane < 16) ? 0 : 8);
            int col = n0 + nf * 16 + rc;
            float v = acc[nf][i] + be[col];
            if (DO_GELU) v = 0.5f * v * (1.0f + erff(v * 0.70710678118f));
            C[(size_t)e * CAPACITY * Ndim + (size_t)row * Ndim + col] = (OUT_T)v;
        }
    }
}

// ---------------------------------------------------------------------------
// Weighted gather/combine back to token order
// ---------------------------------------------------------------------------
__global__ __launch_bounds__(256) void gather_kernel(
    const float* __restrict__ outbuf, const int* __restrict__ topk,
    const int* __restrict__ pos, const float* __restrict__ wgt,
    float* __restrict__ y) {
    int n = blockIdx.x;
    float w0 = wgt[n * 2], w1 = wgt[n * 2 + 1];
    int e0 = topk[n * 2], e1 = topk[n * 2 + 1];
    int p0 = min(pos[n * 2], CAPACITY - 1);
    int p1 = min(pos[n * 2 + 1], CAPACITY - 1);
    const float* s0 = outbuf + ((size_t)e0 * CAPACITY + p0) * D_MODEL;
    const float* s1 = outbuf + ((size_t)e1 * CAPACITY + p1) * D_MODEL;
    for (int d = threadIdx.x; d < D_MODEL; d += blockDim.x)
        y[(size_t)n * D_MODEL + d] = w0 * s0[d] + w1 * s1[d];
}

// ---------------------------------------------------------------------------
extern "C" void kernel_launch(void* const* d_in, const int* in_sizes, int n_in,
                              void* d_out, int out_size, void* d_ws, size_t ws_size,
                              hipStream_t stream) {
    const float* x  = (const float*)d_in[0];
    const float* rw = (const float*)d_in[1];
    const float* rb = (const float*)d_in[2];
    const float* w1 = (const float*)d_in[3];
    const float* b1 = (const float*)d_in[4];
    const float* w2 = (const float*)d_in[5];
    const float* b2 = (const float*)d_in[6];
    float* y = (float*)d_out;

    // workspace layout (256B aligned)
    char* ws = (char*)d_ws;
    size_t off = 0;
    auto alloc = [&](size_t bytes) -> char* {
        char* p = ws + off;
        off = (off + bytes + 255) & ~(size_t)255;
        return p;
    };
    const size_t W1_ELEMS  = (size_t)N_EXPERTS * D_MODEL * D_HIDDEN;
    const size_t W2_ELEMS  = (size_t)N_EXPERTS * D_HIDDEN * D_MODEL;
    const size_t BUF_ELEMS = (size_t)N_EXPERTS * CAPACITY * D_MODEL;
    const size_t H_ELEMS   = (size_t)N_EXPERTS * CAPACITY * D_HIDDEN;

    bf16_t* w1b   = (bf16_t*)alloc(W1_ELEMS * sizeof(bf16_t));
    bf16_t* w2b   = (bf16_t*)alloc(W2_ELEMS * sizeof(bf16_t));
    bf16_t* bufb  = (bf16_t*)alloc(BUF_ELEMS * sizeof(bf16_t));
    bf16_t* hb    = (bf16_t*)alloc(H_ELEMS * sizeof(bf16_t));
    float*  outf  = (float*)alloc(BUF_ELEMS * sizeof(float));
    float*  gates = (float*)alloc((size_t)N_TOKENS * N_EXPERTS * sizeof(float));
    int*    topk  = (int*)alloc((size_t)N_TOKENS * TOP_K * sizeof(int));
    int*    pos   = (int*)alloc((size_t)N_TOKENS * TOP_K * sizeof(int));
    float*  wgt   = (float*)alloc((size_t)N_TOKENS * TOP_K * sizeof(float));
    (void)ws_size;

    cvt_bf16_kernel<<<16384, 256, 0, stream>>>(w1, w1b, W1_ELEMS);
    cvt_bf16_kernel<<<16384, 256, 0, stream>>>(w2, w2b, W2_ELEMS);
    zero_u32_kernel<<<8192, 256, 0, stream>>>((unsigned*)bufb, BUF_ELEMS / 2);
    router_kernel<<<N_TOKENS / 8, 256, 0, stream>>>(x, rw, rb, gates, topk);
    scan_kernel<<<1, 256, 0, stream>>>(topk, gates, pos, wgt);
    dispatch_kernel<<<N_TOKENS * TOP_K, 256, 0, stream>>>(x, topk, pos, bufb);
    {
        dim3 grid(D_HIDDEN / 128, CAPACITY / 128, N_EXPERTS);
        moe_gemm_kernel<D_MODEL, true, bf16_t>
            <<<grid, 256, 0, stream>>>(bufb, w1b, b1, hb, D_HIDDEN);
    }
    {
        dim3 grid(D_MODEL / 128, CAPACITY / 128, N_EXPERTS);
        moe_gemm_kernel<D_HIDDEN, false, float>
            <<<grid, 256, 0, stream>>>(hb, w2b, b2, outf, D_MODEL);
    }
    gather_kernel<<<N_TOKENS, 256, 0, stream>>>(outf, topk, pos, wgt, y);
}
